// Qwen3VLMOETextMOE_44229573214570
// MI455X (gfx1250) — matrix operand order repaired and checked by reading, not compiled, not run on hardware
//
#include <hip/hip_runtime.h>
#include <hip/hip_bf16.h>

// Problem constants (match reference)
#define T_TOK 4096
#define HDIM  2048
#define IDIM  768
#define NEXP  64
#define TOPK  8

#define MTILE 32                        // tokens per ffn block (2 WMMA m-tiles)
#define TILES_PER_EXP (T_TOK / MTILE)   // 128
#define HSTR  (IDIM + 8)                // 776 halfs -> 1552B row stride, bank-conflict free

typedef _Float16 v16h __attribute__((ext_vector_type(16)));
typedef __fp16   v2hr __attribute__((ext_vector_type(2)));   // cvt_pkrtz return type
typedef float    v8f  __attribute__((ext_vector_type(8)));

// Fast SiLU: g * sigmoid(g) with hardware rcp (v_rcp_f32) instead of IEEE div.
__device__ __forceinline__ float fast_silu_mul(float g, float u) {
    return g * __builtin_amdgcn_rcpf(1.f + __expf(-g)) * u;
}

// ---------------------------------------------------------------------------
// Fragment loaders.
// A fragment (16x32 f16): lanes 0-15 hold M=0..15 K=k0+{0..7,16..23},
// lanes 16-31 hold M=0..15 K=k0+{8..15,24..31}.
// ---------------------------------------------------------------------------
__device__ __forceinline__ v16h load_a_f32(const float* __restrict__ row, int k0, int lane) {
    const int kb = k0 + ((lane & 16) ? 8 : 0);
    const float4* p = (const float4*)(row + kb);
    float4 q0 = p[0], q1 = p[1], q2 = p[4], q3 = p[5];   // k+0..7, k+16..23
    union { v16h v; v2hr h[8]; } u;
    u.h[0] = __builtin_amdgcn_cvt_pkrtz(q0.x, q0.y);
    u.h[1] = __builtin_amdgcn_cvt_pkrtz(q0.z, q0.w);
    u.h[2] = __builtin_amdgcn_cvt_pkrtz(q1.x, q1.y);
    u.h[3] = __builtin_amdgcn_cvt_pkrtz(q1.z, q1.w);
    u.h[4] = __builtin_amdgcn_cvt_pkrtz(q2.x, q2.y);
    u.h[5] = __builtin_amdgcn_cvt_pkrtz(q2.z, q2.w);
    u.h[6] = __builtin_amdgcn_cvt_pkrtz(q3.x, q3.y);
    u.h[7] = __builtin_amdgcn_cvt_pkrtz(q3.z, q3.w);
    return u.v;
}

// A fragment from f16 LDS row (Hs): two 16B ds_load_b128 per lane.
__device__ __forceinline__ v16h load_a_h16(const _Float16* __restrict__ row, int k0, int lane) {
    const int kb = k0 + ((lane & 16) ? 8 : 0);
    const uint4* p = (const uint4*)(row + kb);
    union { v16h v; uint4 q[2]; } u;
    u.q[0] = p[0];   // k+0..7
    u.q[1] = p[2];   // k+16..23
    return u.v;
}

// B fragment (32x16 f16) from row-major fp32 weights W[k][n], leading dim ld.
// lanes 0-15: K=k0+{0..7,16..23}, lanes 16-31: K=k0+{8..15,24..31}; N = n0 + lane%16 (col baked
// into `col`). VGPR r holds K = kb+j, kb+j+1 with j = 2r + (r>=4 ? 8 : 0).
__device__ __forceinline__ v16h load_b_f32(const float* __restrict__ col, int k0, int lane, int ld) {
    const int kb = k0 + ((lane & 16) ? 8 : 0);
    const float* p = col + (size_t)kb * (size_t)ld;
    union { v16h v; v2hr h[8]; } u;
#pragma unroll
    for (int r = 0; r < 8; ++r) {
        float f0 = p[(size_t)(2 * r + ((r & 4) ? 8 : 0)) * (size_t)ld];
        float f1 = p[(size_t)(2 * r + 1 + ((r & 4) ? 8 : 0)) * (size_t)ld];
        u.h[r] = __builtin_amdgcn_cvt_pkrtz(f0, f1);
    }
    return u.v;
}

// ---------------------------------------------------------------------------
// Router: logits -> softmax -> top-8 -> renormalize. One 64-thread block/token.
// ---------------------------------------------------------------------------
__global__ __launch_bounds__(64) void router_kernel(const float* __restrict__ x,
                                                    const float* __restrict__ gw,
                                                    int* __restrict__ tki,
                                                    float* __restrict__ tkw) {
    __shared__ __attribute__((aligned(16))) float xrow[HDIM];
    __shared__ float lg[NEXP];
    const int t   = threadIdx.x;
    const int tok = blockIdx.x;

    const float4* xs = (const float4*)(x + (size_t)tok * HDIM);
    for (int i = t; i < HDIM / 4; i += 64) ((float4*)xrow)[i] = xs[i];
    __syncthreads();

    const float* w = gw + (size_t)t * HDIM;
    float acc = 0.f;
    for (int h = 0; h < HDIM; h += 4) {
        float4 wv = *(const float4*)(w + h);
        acc += xrow[h] * wv.x + xrow[h + 1] * wv.y + xrow[h + 2] * wv.z + xrow[h + 3] * wv.w;
    }
    lg[t] = acc;
    __syncthreads();

    if (t == 0) {
        float mx = lg[0];
        for (int i = 1; i < NEXP; ++i) mx = fmaxf(mx, lg[i]);
        for (int i = 0; i < NEXP; ++i) lg[i] = __expf(lg[i] - mx);  // softmax denom cancels in renorm
        int   idxs[TOPK];
        float vals[TOPK];
        float wsum = 0.f;
        for (int j = 0; j < TOPK; ++j) {
            int bi = 0; float bv = -1.f;
            for (int i = 0; i < NEXP; ++i)
                if (lg[i] > bv) { bv = lg[i]; bi = i; }
            idxs[j] = bi; vals[j] = bv; lg[bi] = -2.f; wsum += bv;
        }
        const float inv = 1.f / wsum;
        for (int j = 0; j < TOPK; ++j) {
            tki[tok * TOPK + j] = idxs[j];
            tkw[tok * TOPK + j] = vals[j] * inv;
        }
    }
}

// ---------------------------------------------------------------------------
// Routing bookkeeping
// ---------------------------------------------------------------------------
__global__ void zero_f32_kernel(float* __restrict__ p, int n) {
    int i = blockIdx.x * blockDim.x + threadIdx.x;
    if (i < n) p[i] = 0.f;
}
__global__ void zero_i32_kernel(int* __restrict__ p, int n) {
    int i = blockIdx.x * blockDim.x + threadIdx.x;
    if (i < n) p[i] = 0;
}
__global__ void hist_kernel(const int* __restrict__ tki, int* __restrict__ counts) {
    int i = blockIdx.x * blockDim.x + threadIdx.x;
    if (i < T_TOK * TOPK) atomicAdd(&counts[tki[i]], 1);
}
__global__ void scan_kernel(const int* __restrict__ counts, int* __restrict__ offsets,
                            int* __restrict__ cursor) {
    if (threadIdx.x == 0) {
        int s = 0;
        for (int e = 0; e < NEXP; ++e) { offsets[e] = s; cursor[e] = s; s += counts[e]; }
        offsets[NEXP] = s;
    }
}
__global__ void scatter_kernel(const int* __restrict__ tki, const float* __restrict__ tkw,
                               int* __restrict__ cursor, int* __restrict__ tok_ids,
                               float* __restrict__ tok_w) {
    int i = blockIdx.x * blockDim.x + threadIdx.x;
    if (i < T_TOK * TOPK) {
        int e   = tki[i];
        int pos = atomicAdd(&cursor[e], 1);
        tok_ids[pos] = i / TOPK;
        tok_w[pos]   = tkw[i];
    }
}

// ---------------------------------------------------------------------------
// Grouped expert FFN. grid.x = NEXP * TILES_PER_EXP (same-expert blocks
// adjacent for L2 reuse of the 18.9MB expert weight set). Block = 256 thr
// = 8 waves; 32-token tile = 2 WMMA m-tiles.
// ---------------------------------------------------------------------------
__global__ __launch_bounds__(256) void moe_ffn_kernel(const float* __restrict__ x,
                                                      const float* __restrict__ wgu,
                                                      const float* __restrict__ wd,
                                                      const int* __restrict__ tok_ids,
                                                      const float* __restrict__ tok_w,
                                                      const int* __restrict__ offsets,
                                                      float* __restrict__ out) {
    __shared__ __attribute__((aligned(16))) _Float16 Hs[MTILE * HSTR];
    __shared__ int   tid_s[MTILE];
    __shared__ float tw_s[MTILE];

    const int e    = blockIdx.x >> 7;        // / TILES_PER_EXP
    const int tile = blockIdx.x & (TILES_PER_EXP - 1);
    const int off0  = offsets[e];
    const int nrows = offsets[e + 1] - off0;
    const int row0  = tile * MTILE;
    if (row0 >= nrows) return;

    const int t = threadIdx.x;
    if (t < MTILE) {
        int r = row0 + t;
        if (r < nrows) { tid_s[t] = tok_ids[off0 + r]; tw_s[t] = tok_w[off0 + r]; }
        else           { tid_s[t] = tok_ids[off0];     tw_s[t] = 0.f; }  // padded row, weight 0
    }
    __syncthreads();

    const float* Wgu_e = wgu + (size_t)e * HDIM * (2 * IDIM);
    const float* Wd_e  = wd  + (size_t)e * IDIM * HDIM;
    __builtin_prefetch(Wgu_e + ((size_t)t << 8), 0, 0);   // global_prefetch_b8 warm-up

    const int wave = t >> 5;
    const int lane = t & 31;
    const int l15  = lane & 15;
    const int hi   = lane >> 4;

    const float* arow0 = x + (size_t)tid_s[l15]      * HDIM;  // m-tile 0, M = l15
    const float* arow1 = x + (size_t)tid_s[l15 + 16] * HDIM;  // m-tile 1, M = l15+16

    // ---- GEMM1: gu = X @ Wgu, fused SiLU(gate)*up -> Hs (f16) -------------
    for (int p = wave; p < IDIM / 16; p += 8) {              // 48 gate/up column pairs
        v8f ag0 = {}, ag1 = {}, au0 = {}, au1 = {};
        const float* colg = Wgu_e + p * 16 + l15;            // gate cols [0, I)
        const float* colu = Wgu_e + IDIM + p * 16 + l15;     // up   cols [I, 2I)
        for (int k0 = 0; k0 < HDIM; k0 += 32) {
            v16h A0 = load_a_f32(arow0, k0, lane);
            v16h A1 = load_a_f32(arow1, k0, lane);
            v16h Bg = load_b_f32(colg, k0, lane, 2 * IDIM);
            v16h Bu = load_b_f32(colu, k0, lane, 2 * IDIM);
            ag0 = __builtin_amdgcn_wmma_f32_16x16x32_f16(false, A0, false, Bg, (short)0, ag0, false, false);
            ag1 = __builtin_amdgcn_wmma_f32_16x16x32_f16(false, A1, false, Bg, (short)0, ag1, false, false);
            au0 = __builtin_amdgcn_wmma_f32_16x16x32_f16(false, A0, false, Bu, (short)0, au0, false, false);
            au1 = __builtin_amdgcn_wmma_f32_16x16x32_f16(false, A1, false, Bu, (short)0, au1, false, false);
        }
        const int nc = p * 16 + l15;
#pragma unroll
        for (int r = 0; r < 8; ++r) {
            // C layout: lane<16 -> M=r, lane>=16 -> M=r+8; N = lane%16
            const int m0 = r + 8 * hi;
            Hs[(size_t)m0 * HSTR + nc]        = (_Float16)fast_silu_mul(ag0[r], au0[r]);
            Hs[(size_t)(m0 + 16) * HSTR + nc] = (_Float16)fast_silu_mul(ag1[r], au1[r]);
        }
    }
    __syncthreads();

    // ---- GEMM2: out += (Hs @ Wd) * route_weight ---------------------------
    const _Float16* hrow0 = Hs + (size_t)l15 * HSTR;
    const _Float16* hrow1 = Hs + (size_t)(l15 + 16) * HSTR;
    for (int nt = wave; nt < HDIM / 16; nt += 8) {           // 128 output n-tiles
        v8f c0 = {}, c1 = {};
        const float* col = Wd_e + nt * 16 + l15;
        for (int k0 = 0; k0 < IDIM; k0 += 32) {
            v16h A0 = load_a_h16(hrow0, k0, lane);
            v16h A1 = load_a_h16(hrow1, k0, lane);
            v16h B  = load_b_f32(col, k0, lane, HDIM);
            c0 = __builtin_amdgcn_wmma_f32_16x16x32_f16(false, A0, false, B, (short)0, c0, false, false);
            c1 = __builtin_amdgcn_wmma_f32_16x16x32_f16(false, A1, false, B, (short)0, c1, false, false);
        }
        const int nc = nt * 16 + l15;
#pragma unroll
        for (int r = 0; r < 8; ++r) {
            const int m0 = r + 8 * hi;
            atomicAdd(out + (size_t)tid_s[m0] * HDIM + nc,      c0[r] * tw_s[m0]);
            atomicAdd(out + (size_t)tid_s[m0 + 16] * HDIM + nc, c1[r] * tw_s[m0 + 16]);
        }
    }
}

// ---------------------------------------------------------------------------
extern "C" void kernel_launch(void* const* d_in, const int* in_sizes, int n_in,
                              void* d_out, int out_size, void* d_ws, size_t ws_size,
                              hipStream_t stream) {
    (void)in_sizes; (void)n_in; (void)out_size; (void)ws_size;
    const float* x   = (const float*)d_in[0];
    const float* gw  = (const float*)d_in[1];
    const float* wgu = (const float*)d_in[2];
    const float* wd  = (const float*)d_in[3];
    // d_in[4] = top_k scalar; fixed at 8 here.
    float* out = (float*)d_out;

    char* ws = (char*)d_ws;
    int*   tki      = (int*)ws;   ws += (size_t)T_TOK * TOPK * 4;
    float* tkw      = (float*)ws; ws += (size_t)T_TOK * TOPK * 4;
    int*   counts   = (int*)ws;   ws += 256;
    int*   offsets  = (int*)ws;   ws += 512;
    int*   cursor   = (int*)ws;   ws += 256;
    int*   tok_ids  = (int*)ws;   ws += (size_t)T_TOK * TOPK * 4;
    float* tok_w    = (float*)ws; ws += (size_t)T_TOK * TOPK * 4;

    const int nout = T_TOK * HDIM;
    zero_f32_kernel<<<(nout + 255) / 256, 256, 0, stream>>>(out, nout);
    zero_i32_kernel<<<1, 64, 0, stream>>>(counts, NEXP);

    router_kernel<<<T_TOK, 64, 0, stream>>>(x, gw, tki, tkw);

    const int npairs = T_TOK * TOPK;
    hist_kernel<<<(npairs + 255) / 256, 256, 0, stream>>>(tki, counts);
    scan_kernel<<<1, 64, 0, stream>>>(counts, offsets, cursor);
    scatter_kernel<<<(npairs + 255) / 256, 256, 0, stream>>>(tki, tkw, cursor, tok_ids, tok_w);

    moe_ffn_kernel<<<NEXP * TILES_PER_EXP, 256, 0, stream>>>(x, wgu, wd, tok_ids, tok_w, offsets, out);
}
